// BinaryWeightConv2d_1391569404413
// MI455X (gfx1250) — matrix operand order repaired and checked by reading, not compiled, not run on hardware
//
#include <hip/hip_runtime.h>
#include <hip/hip_bf16.h>

typedef __attribute__((ext_vector_type(16))) _Float16 v16h;
typedef __attribute__((ext_vector_type(8)))  float    v8f;
typedef unsigned int u32x4 __attribute__((ext_vector_type(4)));
typedef int          i32x8 __attribute__((ext_vector_type(8)));
typedef int          i32x4 __attribute__((ext_vector_type(4)));

#define NBATCH 32
#define CIN    128
#define COUT   256
#define HW     56
#define IMG    (HW*HW)          // 3136
#define KTOT   1152             // 128*3*3
#define KCH    36               // K chunks of 32
#define OCT    16               // 256/16 oc tiles
#define OCPW   4                // oc tiles per wave (64 output channels)
#define WPB    8                // waves per block (8 pixel tiles)
#define NTILES (NBATCH*IMG/16)  // 6272 pixel tiles
#define A_HALFS      ((size_t)OCT*KCH*32*16)       // 294912 f16 = 589824 B
#define A_SLICE_B    (OCPW*KCH*32*32)              // 147456 B per oc-group
#define A_SLICE_DW   (A_SLICE_B/4)                 // 36864 dwords

// ---------------- Kernel 1: per-output-channel scale = clamp(mean|w|, 1e-8) ----
__global__ __launch_bounds__(32) void bconv_scale_kernel(const float* __restrict__ w,
                                                         float* __restrict__ scale) {
    int oc   = blockIdx.x;          // 256 blocks, one wave each
    int lane = threadIdx.x;         // 32 lanes
    float s = 0.0f;
    #pragma unroll
    for (int i = lane; i < KTOT; i += 32)
        s += fabsf(w[(size_t)oc * KTOT + i]);
    #pragma unroll
    for (int off = 16; off > 0; off >>= 1)
        s += __shfl_xor(s, off, 32);
    if (lane == 0)
        scale[oc] = fmaxf(s * (1.0f / (float)KTOT), 1e-8f);
}

// ---------------- Kernel 2: pack sign(w) into WMMA-A f16 layout ----------------
// Apack element index: ((ocTile*36 + kc)*32 + lane)*16 + i
// A 16x32 f16 layout (wave32): M = lane&15, K(i) = (i<8 ? i : i+8) + 8*(lane>>4)
// K ordering: k = (kh*3+kw)*128 + ic  ->  within chunk kc: (kh,kw) fixed = kc>>2,
// ic = (kc&3)*32 + K(i)
__global__ __launch_bounds__(256) void bconv_pack_kernel(const float* __restrict__ w,
                                                         _Float16* __restrict__ Apack) {
    int t = blockIdx.x * blockDim.x + threadIdx.x;
    if (t >= OCT * KCH * 32) return;
    int lane   = t & 31;
    int kc     = (t >> 5) % KCH;
    int ocTile = t / (KCH * 32);
    int oc     = ocTile * 16 + (lane & 15);
    int half   = lane >> 4;
    int kk     = kc >> 2;            // 0..8
    int kh     = kk / 3;
    int kw     = kk - kh * 3;
    int icc    = (kc & 3) * 32;
    _Float16* dst = Apack + (size_t)t * 16;
    #pragma unroll
    for (int i = 0; i < 16; ++i) {
        int kin = (i < 8 ? i : i + 8) + 8 * half;
        int ic  = icc + kin;
        float wv = w[(size_t)oc * KTOT + (size_t)ic * 9 + kh * 3 + kw];
        dst[i] = (_Float16)(wv > 0.0f ? 1.0f : (wv < 0.0f ? -1.0f : 0.0f));
    }
}

// ---------------- Kernel 3: implicit-GEMM binary conv via WMMA + TDM -----------
// Block = 8 waves (8 pixel tiles) x one 64-oc group. The block's 144KB packed-A
// slice is DMA'd into LDS once by the Tensor Data Mover, then every WMMA reads
// A via ds_load. Each B tile (16 f32 loads + cvt) feeds 4 v_wmma ops.
__global__ __launch_bounds__(WPB*32) void bconv_wmma_kernel(const float* __restrict__ x,
                                                            const _Float16* __restrict__ Apack,
                                                            const float* __restrict__ scale,
                                                            const float* __restrict__ bias,
                                                            float* __restrict__ out) {
    extern __shared__ _Float16 smemA[];   // A_SLICE_B bytes, dynamic -> LDS offset 0

    const int lane    = threadIdx.x & 31;
    const int wave    = threadIdx.x >> 5;
    const int tileN   = blockIdx.x * WPB + wave;  // 6272 pixel tiles / 8 waves
    const int tileOC0 = blockIdx.y * OCPW;        // 4 oc-tile group -> 64 oc
    const int ncol    = lane & 15;
    const int half    = lane >> 4;

    // ---- TDM: load this block's packed-A slice (contiguous) into LDS ----
    if (threadIdx.x == 0) {
        unsigned long long ga =
            (unsigned long long)(const char*)(Apack + (size_t)tileOC0 * KCH * 32 * 16);
        u32x4 g0;
        g0[0] = 1u;                                   // count=1 (valid descriptor)
        g0[1] = 0u;                                   // lds_addr = 0 (dynamic LDS base)
        g0[2] = (unsigned)(ga & 0xFFFFFFFFu);         // global_addr[31:0]
        g0[3] = (unsigned)((ga >> 32) & 0x01FFFFFFu)  // global_addr[56:32]
              | (2u << 30);                           // type = 2 ("image")
        i32x8 g1;
        const unsigned e = A_SLICE_DW;                // 36864 dword elements
        g1[0] = 0x00020000;                           // data_size=2 (4B), mask=0
        g1[1] = (int)((e & 0xFFFFu) << 16);           // tensor_dim0[15:0]
        g1[2] = (int)(((e >> 16) & 0xFFFFu) | (1u << 16)); // dim0[31:16], tensor_dim1=1
        g1[3] = (int)((e & 0xFFFFu) << 16);           // tile_dim0 = e
        g1[4] = 1;                                    // tile_dim1 = 1, tile_dim2 = 0
        g1[5] = (int)e;                               // tensor_dim0_stride[31:0]
        g1[6] = 0;                                    // stride0[47:32], stride1[15:0]
        g1[7] = 0;
        i32x4 gz4 = {0, 0, 0, 0};                     // groups 2/3 unused (<=2D)
        i32x8 gz8 = {0, 0, 0, 0, 0, 0, 0, 0};        // extra group (6-arg toolchain)
        __builtin_amdgcn_tensor_load_to_lds(g0, g1, gz4, gz4, gz8, 0);
        __builtin_amdgcn_s_wait_tensorcnt((short)0);
    }
    __syncthreads();

    // decode pixel n -> (b, oh, ow)
    const int n   = tileN * 16 + ncol;
    const int b_  = n / IMG;
    const int rem = n - b_ * IMG;
    const int oh  = rem / 56;
    const int ow  = rem - oh * 56;

    v8f acc[OCPW];
    #pragma unroll
    for (int t = 0; t < OCPW; ++t) acc[t] = (v8f){};

    const v16h* __restrict__ sA = (const v16h*)smemA;   // [(t*KCH+kc)*32 + lane]

    #pragma unroll 4
    for (int kc = 0; kc < KCH; ++kc) {
        const int kk = kc >> 2;          // (kh,kw) — constant per unroll group
        const int kh = kk / 3;
        const int kw = kk - kh * 3;
        const int ih = oh + kh - 1;
        const int iw = ow + kw - 1;
        const bool valid = ((unsigned)ih < (unsigned)HW) & ((unsigned)iw < (unsigned)HW);

        // B 32x16 f16 layout: N = lane&15, K(i) = (i<8 ? i : i+8) + 8*half
        const int icbase = (kc & 3) * 32 + half * 8;
        const float* __restrict__ xp =
            x + ((size_t)(b_ * CIN + icbase) * IMG + (size_t)ih * HW + iw);

        v16h bv;
        #pragma unroll
        for (int i = 0; i < 16; ++i) {
            float xv = valid ? xp[(size_t)(i < 8 ? i : i + 8) * IMG] : 0.0f;
            bv[i] = (_Float16)xv;
        }

        // 4 packed-A tiles from LDS (ds_load), 4 WMMAs sharing bv
        #pragma unroll
        for (int t = 0; t < OCPW; ++t) {
            v16h av = sA[((size_t)t * KCH + kc) * 32 + lane];
            acc[t] = __builtin_amdgcn_wmma_f32_16x16x32_f16(
                /*neg_a=*/false, av, /*neg_b=*/false, bv,
                /*c_mod=*/(short)0, acc[t], /*reuse_a=*/false, /*reuse_b=*/false);
        }
    }

    // Epilogue: scale + bias, ternarize, store.
    // C/D layout: VGPR r holds M = r + 8*half, lane&15 = N
    #pragma unroll
    for (int t = 0; t < OCPW; ++t) {
        #pragma unroll
        for (int r = 0; r < 8; ++r) {
            int m  = r + half * 8;
            int oc = (tileOC0 + t) * 16 + m;
            float v  = acc[t][r] * scale[oc] + bias[oc];
            float tz = v > 0.5f ? 1.0f : (v < -0.5f ? -1.0f : 0.0f);
            out[((size_t)b_ * COUT + oc) * IMG + (size_t)oh * HW + ow] = tz;
        }
    }
}

extern "C" void kernel_launch(void* const* d_in, const int* in_sizes, int n_in,
                              void* d_out, int out_size, void* d_ws, size_t ws_size,
                              hipStream_t stream) {
    const float* x    = (const float*)d_in[0];
    const float* w    = (const float*)d_in[1];
    const float* bias = (const float*)d_in[2];
    float*       out  = (float*)d_out;

    _Float16* Apack = (_Float16*)d_ws;
    float*    scale = (float*)((char*)d_ws + A_HALFS * sizeof(_Float16));

    // 1) per-oc scale
    bconv_scale_kernel<<<COUT, 32, 0, stream>>>(w, scale);
    // 2) pack binarized weights into WMMA-A layout
    {
        int threads = OCT * KCH * 32;   // 18432
        bconv_pack_kernel<<<(threads + 255) / 256, 256, 0, stream>>>(w, Apack);
    }
    // 3) implicit GEMM conv: 8 pixel tiles x 64 oc per block, A via TDM->LDS
    {
        dim3 grid(NTILES / WPB, OCT / OCPW);   // (784, 4)
        bconv_wmma_kernel<<<grid, WPB * 32, A_SLICE_B, stream>>>(x, Apack, scale, bias, out);
    }
}
